// SHMAP_25091198943741
// MI455X (gfx1250) — matrix operand order repaired and checked by reading, not compiled, not run on hardware
//
#include <hip/hip_runtime.h>

typedef __attribute__((ext_vector_type(16))) _Float16 v16h;
typedef __attribute__((ext_vector_type(8)))  _Float16 v8h;
typedef __attribute__((ext_vector_type(4)))  _Float16 v4h;
typedef __attribute__((ext_vector_type(8)))  float    v8f;

#define DIM 512
#define KB  4
#define NH  8
#define BB  16
#define TT  1024
#define TSS 512
#define NQ  4
#define NT  32
#define MATH (512*512)   // halves per packed weight matrix

static __device__ __forceinline__ v8f wmma_f16(v16h a, v16h b, v8f c) {
  // D = A(16x32 f16) * B(32x16 f16) + C(16x16 f32)
  return __builtin_amdgcn_wmma_f32_16x16x32_f16(false, a, false, b, (short)0, c,
                                                false, false);
}

// Async copy 16B global -> LDS (ASYNCcnt-tracked, no VGPR data path).
// lds_addr is the 32-bit LDS byte address (low 32 bits of the generic pointer).
static __device__ __forceinline__ void async_ld_b128(unsigned lds_addr, const void* g) {
  asm volatile("global_load_async_to_lds_b128 %0, %1, off"
               :: "v"(lds_addr), "v"((unsigned long long)(size_t)g)
               : "memory");
}
static __device__ __forceinline__ void wait_async0() {
  asm volatile("s_wait_asynccnt 0x0" ::: "memory");
}

// ---------------------------------------------------------------------------
// Pack an O x C (512x512) fp32 matrix into WMMA A-fragment order, f16.
// Fragment mapping used here (contiguous k, not the ISA table order; 16-bit A 16x32): lane L -> M = L&15,
// half h -> K = 16*(L>>4) + h.
// Packed element order: [mtile(32)][kc(16)][lane(32)][h(16)]
// ---------------------------------------------------------------------------
__global__ void k_pack(const float* __restrict__ src, int o_stride, int c_stride,
                       _Float16* __restrict__ dst) {
  int e = blockIdx.x * blockDim.x + threadIdx.x;   // 0 .. 512*512-1
  int h     = e & 15;
  int lane  = (e >> 4) & 31;
  int kc    = (e >> 9) & 15;
  int mtile = e >> 13;
  int o = mtile * 16 + (lane & 15);
  int c = kc * 32 + ((lane >> 4) << 4) + h;
  dst[e] = (_Float16)src[(size_t)o * o_stride + (size_t)c * c_stride];
}

// ---------------------------------------------------------------------------
// Fold BN (scale = g/sqrt(1+eps)) and conv bias into per-channel alpha/beta.
// bn layout (floats): [0]=proj1 alpha 3*512, [1536]=proj1 beta,
// [3072]=ffn1 alpha 4*512, [5120]=ffn1 beta, [7168]=ffn2 alpha, [9216]=ffn2 beta
// ---------------------------------------------------------------------------
__global__ void k_prep_bn(const float* __restrict__ g1,  const float* __restrict__ be1,
                          const float* __restrict__ b1,
                          const float* __restrict__ fg1, const float* __restrict__ fbe1,
                          const float* __restrict__ fb1,
                          const float* __restrict__ fg2, const float* __restrict__ fbe2,
                          const float* __restrict__ fb2,
                          float* __restrict__ bn) {
  int id = blockIdx.x * blockDim.x + threadIdx.x;
  const float r = rsqrtf(1.0f + 1e-5f);
  if (id < 1536) {
    float sc = g1[id] * r;
    bn[id] = sc;
    bn[1536 + id] = sc * b1[id] + be1[id];
  } else if (id < 1536 + 2048) {
    int j = id - 1536;
    float sc = fg1[j] * r;
    bn[3072 + j] = sc;
    bn[5120 + j] = sc * fb1[j] + fbe1[j];
  } else if (id < 1536 + 4096) {
    int j = id - 3584;
    float sc = fg2[j] * r;
    bn[7168 + j] = sc;
    bn[9216 + j] = sc * fb2[j] + fbe2[j];
  }
}

// ---------------------------------------------------------------------------
// Fused proj1 (3 branches): h = relu(alpha*(W1@f)+beta) kept only in WMMA
// accumulators; logits = W2@h accumulated per-lane then via LDS atomics;
// softmax across the 3 branches -> att (B,3,KB,T).
// Block: 256 thr (8 waves); wave w owns O rows [64w,64w+64); tile N = 32.
// ---------------------------------------------------------------------------
__global__ void __launch_bounds__(256)
k_proj1_att(const float* __restrict__ f,        // (3,B,DIM,T)
            const _Float16* __restrict__ packA, // matrices 0..2
            const float* __restrict__ w2,       // (3,KB,DIM)
            const float* __restrict__ b2,       // (3,KB)
            const float* __restrict__ bn,
            float* __restrict__ att)            // (B,3,KB,T)
{
  __shared__ __align__(32) _Float16 ldsB[NT * 32];
  __shared__ float lg[3][KB][NT];
  const int tid  = threadIdx.x;
  const int lane = tid & 31, wave = tid >> 5;
  const int ntile = blockIdx.x;
  const int b  = ntile >> 5;            // 32 tiles per batch row
  const int t0 = (ntile & 31) * NT;
  const int scc = tid >> 3;             // staging: channel 0..31
  const int st4 = (tid & 7) * 4;        // staging: t offset 0,4,..,28

  for (int i = 0; i < 3; ++i) {
    if (tid < KB * NT) lg[i][tid >> 5][tid & 31] = 0.0f;
    v8f acc[4][2];
    const v8f vz = {0.f,0.f,0.f,0.f,0.f,0.f,0.f,0.f};
#pragma unroll
    for (int m = 0; m < 4; ++m) { acc[m][0] = vz; acc[m][1] = vz; }
    __syncthreads();

    for (int kc = 0; kc < 16; ++kc) {
      // stage 32c x 32t chunk -> LDS [t][c] as f16 (one b128 per thread)
      {
        const float4 v4 = *(const float4*)&f[(((size_t)i * BB + b) * DIM +
                                              kc * 32 + scc) * TT + t0 + st4];
        ldsB[(st4 + 0) * 32 + scc] = (_Float16)v4.x;
        ldsB[(st4 + 1) * 32 + scc] = (_Float16)v4.y;
        ldsB[(st4 + 2) * 32 + scc] = (_Float16)v4.z;
        ldsB[(st4 + 3) * 32 + scc] = (_Float16)v4.w;
      }
      __syncthreads();
      v16h bf[2];
#pragma unroll
      for (int nt = 0; nt < 2; ++nt) {
        int base = (nt * 16 + (lane & 15)) * 32 + ((lane >> 4) << 4);
        bf[nt] = *(const v16h*)&ldsB[base];
      }
#pragma unroll
      for (int m = 0; m < 4; ++m) {
        int mtile = wave * 4 + m;
        const v16h af = *(const v16h*)&packA[(size_t)i * MATH +
                          (((size_t)mtile * 16 + kc) * 32 + lane) * 16];
        acc[m][0] = wmma_f16(af, bf[0], acc[m][0]);
        acc[m][1] = wmma_f16(af, bf[1], acc[m][1]);
      }
      __syncthreads();
    }

    // epilogue: bn+relu, accumulate logits (sum over this wave's O rows)
    float part[2][KB] = {};
#pragma unroll
    for (int m = 0; m < 4; ++m) {
#pragma unroll
      for (int r = 0; r < 8; ++r) {
        int o = (wave * 4 + m) * 16 + r + ((lane >> 4) << 3);
        float al = bn[i * DIM + o];
        float be = bn[1536 + i * DIM + o];
        float w2v[KB];
#pragma unroll
        for (int k = 0; k < KB; ++k) w2v[k] = w2[((size_t)i * KB + k) * DIM + o];
#pragma unroll
        for (int nt = 0; nt < 2; ++nt) {
          float v = acc[m][nt][r] * al + be;
          v = v > 0.f ? v : 0.f;
#pragma unroll
          for (int k = 0; k < KB; ++k) part[nt][k] += w2v[k] * v;
        }
      }
    }
#pragma unroll
    for (int nt = 0; nt < 2; ++nt)
#pragma unroll
      for (int k = 0; k < KB; ++k)
        atomicAdd(&lg[i][k][nt * 16 + (lane & 15)], part[nt][k]);
    __syncthreads();
  }

  // 3-way softmax across branches
  if (tid < KB * NT) {
    int k = tid >> 5, n = tid & 31;
    float l0 = lg[0][k][n] + b2[0 * KB + k];
    float l1 = lg[1][k][n] + b2[1 * KB + k];
    float l2 = lg[2][k][n] + b2[2 * KB + k];
    float mx = fmaxf(l0, fmaxf(l1, l2));
    float e0 = __expf(l0 - mx), e1 = __expf(l1 - mx), e2 = __expf(l2 - mx);
    float inv = 1.0f / (e0 + e1 + e2);
    size_t base = (((size_t)b * 3 + 0) * KB + k) * TT + t0 + n;
    att[base]                         = e0 * inv;
    att[base + (size_t)KB * TT]       = e1 * inv;
    att[base + (size_t)2 * KB * TT]   = e2 * inv;
  }
}

// ---------------------------------------------------------------------------
// nk[k] = sum_i att[b,i,k,t]*f[i,b,c,t]  (f16 out), plus pooled residual
// written (store, initializes nf).  Block: (cb,q,b), 32 c x 8 t-groups.
// att tile staged to LDS with GLOBAL_LOAD_ASYNC_TO_LDS_B128 (ASYNCcnt).
// ---------------------------------------------------------------------------
__global__ void __launch_bounds__(256)
k_nk(const float* __restrict__ f, const float* __restrict__ att,
     _Float16* __restrict__ nk, float* __restrict__ nf)
{
  __shared__ __align__(16) float attS[3 * KB][256];
  __shared__ float red[KB][32][8];
  const int tid = threadIdx.x;
  const int cb = blockIdx.x, q = blockIdx.y, b = blockIdx.z;

  // async-stage 12 rows x 256 floats of att (contiguous -> contiguous)
  {
    const unsigned ldsBase = (unsigned)(size_t)&attS[0][0];
#pragma unroll
    for (int it = 0; it < 3; ++it) {
      int ev = tid + it * 256;                  // 768 x 16B vectors
      int ik = ev >> 6;
      int toff = (ev & 63) * 4;
      const float* g = att + ((size_t)b * 3 * KB + ik) * TT + q * 256 + toff;
      async_ld_b128(ldsBase + (unsigned)ev * 16u, g);
    }
    wait_async0();
  }
  __syncthreads();

  const int cl = tid >> 3, tg = tid & 7;
  const int c = cb * 32 + cl;
  float sum[KB] = {};
  const size_t fbase = (((size_t)0 * BB + b) * DIM + c) * TT + q * 256 + tg * 32;
  const size_t fstep = (size_t)BB * DIM * TT;
  for (int j0 = 0; j0 < 32; j0 += 4) {
    const float4 a0 = *(const float4*)&f[fbase + j0];
    const float4 a1 = *(const float4*)&f[fbase + fstep + j0];
    const float4 a2 = *(const float4*)&f[fbase + 2 * fstep + j0];
    const float f0a[4] = {a0.x, a0.y, a0.z, a0.w};
    const float f1a[4] = {a1.x, a1.y, a1.z, a1.w};
    const float f2a[4] = {a2.x, a2.y, a2.z, a2.w};
#pragma unroll
    for (int k = 0; k < KB; ++k) {
      v4h ov;
#pragma unroll
      for (int j = 0; j < 4; ++j) {
        int tloc = tg * 32 + j0 + j;
        float v = attS[0 * KB + k][tloc] * f0a[j] +
                  attS[1 * KB + k][tloc] * f1a[j] +
                  attS[2 * KB + k][tloc] * f2a[j];
        ov[j] = (_Float16)v;
        sum[k] += v;
      }
      *(v4h*)&nk[(((size_t)k * BB + b) * DIM + c) * TT + q * 256 + tg * 32 + j0] = ov;
    }
  }
#pragma unroll
  for (int k = 0; k < KB; ++k) red[k][cl][tg] = sum[k];
  __syncthreads();
  if (tid < KB * 32) {
    int k = tid >> 5, cl2 = tid & 31;
    float s = 0.f;
#pragma unroll
    for (int g = 0; g < 8; ++g) s += red[k][cl2][g];
    nf[(((size_t)b * NQ + q) * KB + k) * DIM + cb * 32 + cl2] = s * (1.0f / 256.0f);
  }
}

// ---------------------------------------------------------------------------
// conv3 (SAME, ksize 3) as 3 accumulated WMMA GEMMs over tap-shifted LDS tiles.
// mode 0: write relu(bn(.)) as f16 to outH.
// mode 1: relu(bn(.)) then pooled accumulation into nf (shfl reduce + atomic).
// ---------------------------------------------------------------------------
__global__ void __launch_bounds__(256)
k_conv3(const _Float16* __restrict__ in,    // (KB,B,DIM,T) f16
        const _Float16* __restrict__ packA,
        int matBase, const float* __restrict__ bn, int bnOff,
        _Float16* __restrict__ outH, float* __restrict__ nf, int mode)
{
  __shared__ __align__(32) _Float16 ldsB[34 * 32];
  const int tid  = threadIdx.x;
  const int lane = tid & 31, wave = tid >> 5;
  const int ntile = blockIdx.x;
  const int k = blockIdx.y;
  const int b  = ntile >> 5;
  const int t0 = (ntile & 31) * NT;
  const size_t inBase = (((size_t)k * BB + b) * DIM) * TT;

  v8f acc[4][2];
  const v8f vz = {0.f,0.f,0.f,0.f,0.f,0.f,0.f,0.f};
#pragma unroll
  for (int m = 0; m < 4; ++m) { acc[m][0] = vz; acc[m][1] = vz; }

  for (int kc = 0; kc < 16; ++kc) {
    __syncthreads();
    // stage 34 t rows (t0-1 .. t0+32) x 32 c, zero-padded at sequence ends
    for (int e = tid; e < 34 * 32; e += 256) {
      int ttl = e % 34, cc = e / 34;
      int t = t0 - 1 + ttl;
      _Float16 v = (_Float16)0.f;
      if (t >= 0 && t < TT) v = in[inBase + (size_t)(kc * 32 + cc) * TT + t];
      ldsB[ttl * 32 + cc] = v;
    }
    if (kc + 1 < 16 && tid < 32)
      __builtin_prefetch(&in[inBase + (size_t)((kc + 1) * 32 + tid) * TT + t0], 0, 1);
    __syncthreads();

#pragma unroll
    for (int tap = 0; tap < 3; ++tap) {
      v16h bf[2];
#pragma unroll
      for (int nt = 0; nt < 2; ++nt) {
        int base = (nt * 16 + (lane & 15) + tap) * 32 + ((lane >> 4) << 4);
        bf[nt] = *(const v16h*)&ldsB[base];
      }
#pragma unroll
      for (int m = 0; m < 4; ++m) {
        int mtile = wave * 4 + m;
        int mat = matBase + k * 3 + tap;
        const v16h af = *(const v16h*)&packA[(size_t)mat * MATH +
                          (((size_t)mtile * 16 + kc) * 32 + lane) * 16];
        acc[m][0] = wmma_f16(af, bf[0], acc[m][0]);
        acc[m][1] = wmma_f16(af, bf[1], acc[m][1]);
      }
    }
  }

  const float* al = bn + bnOff + k * DIM;
  const float* be = bn + bnOff + 2048 + k * DIM;
  if (mode == 0) {
#pragma unroll
    for (int m = 0; m < 4; ++m)
#pragma unroll
      for (int r = 0; r < 8; ++r) {
        int o = (wave * 4 + m) * 16 + r + ((lane >> 4) << 3);
        float a = al[o], bb = be[o];
#pragma unroll
        for (int nt = 0; nt < 2; ++nt) {
          float v = acc[m][nt][r] * a + bb;
          v = v > 0.f ? v : 0.f;
          outH[inBase + (size_t)o * TT + t0 + nt * 16 + (lane & 15)] = (_Float16)v;
        }
      }
  } else {
    const int q = t0 >> 8;
#pragma unroll
    for (int m = 0; m < 4; ++m)
#pragma unroll
      for (int r = 0; r < 8; ++r) {
        int o = (wave * 4 + m) * 16 + r + ((lane >> 4) << 3);
        float a = al[o], bb = be[o];
        float s = 0.f;
#pragma unroll
        for (int nt = 0; nt < 2; ++nt) {
          float v = acc[m][nt][r] * a + bb;
          s += (v > 0.f ? v : 0.f);
        }
        s += __shfl_xor(s, 1, 32);
        s += __shfl_xor(s, 2, 32);
        s += __shfl_xor(s, 4, 32);
        s += __shfl_xor(s, 8, 32);
        if ((lane & 15) == 0)
          atomicAdd(&nf[(((size_t)b * NQ + q) * KB + k) * DIM + o], s * (1.0f / 256.0f));
      }
  }
}

// ---------------------------------------------------------------------------
// Pool-before-GEMM helpers (proj2 and policy are linear, pooling commutes).
// ---------------------------------------------------------------------------
__global__ void __launch_bounds__(256)
k_pool_cs(const float* __restrict__ cin, const float* __restrict__ sin,
          float* __restrict__ P) {
  int bq = blockIdx.x; int b = bq >> 2, q = bq & 3;
  for (int cc = threadIdx.x; cc < 1024; cc += 256) {
    float s = 0.f;
    if (cc < 512) {
      const float* p = cin + ((size_t)b * DIM + cc) * TT + q * 256;
      for (int t = 0; t < 256; ++t) s += p[t];
      s *= (1.0f / 256.0f);
    } else {
      // s2[t] = s[t>>1] -> mean over 128 source samples
      const float* p = sin + ((size_t)b * DIM + (cc - 512)) * TSS + q * 128;
      for (int t = 0; t < 128; ++t) s += p[t];
      s *= (1.0f / 128.0f);
    }
    P[((size_t)b * 1024 + cc) * NQ + q] = s;
  }
}

__global__ void __launch_bounds__(256)
k_pool_f(const float* __restrict__ f, float* __restrict__ Pf) {
  int bq = blockIdx.x; int b = bq >> 2, q = bq & 3;
  for (int j = threadIdx.x; j < 1536; j += 256) {
    int i = j >> 9, c = j & 511;
    const float* p = f + (((size_t)i * BB + b) * DIM + c) * TT + q * 256;
    float s = 0.f;
    for (int t = 0; t < 256; ++t) s += p[t];
    Pf[((size_t)b * 1536 + j) * NQ + q] = s * (1.0f / 256.0f);
  }
}

// ---------------------------------------------------------------------------
// Final: policy+gumbel hard selection (ip), q/K/V projections, 4-key
// masked attention, outputs (out, ip). One block per (b,q).
// ---------------------------------------------------------------------------
__global__ void __launch_bounds__(256)
k_final(const float* __restrict__ Pcs, const float* __restrict__ Pf,
        const float* __restrict__ nf,
        const float* __restrict__ proj2_w, const float* __restrict__ proj2_b,
        const float* __restrict__ policy_w, const float* __restrict__ policy_b,
        const float* __restrict__ qkv_w, const float* __restrict__ qkv_b,
        const float* __restrict__ gu, const int* __restrict__ tau_p,
        float* __restrict__ out)
{
  __shared__ float qv[DIM], Qh[DIM], Kh[KB * DIM], Vh[KB * DIM];
  __shared__ float sc[NH * KB], at[NH * KB], maskS[KB], plA[KB];
  const int tid = threadIdx.x;
  const int bq = blockIdx.x; const int b = bq >> 2, q = bq & 3;
  const float tau = (float)tau_p[0];

  if (tid < KB) {
    int k = tid;
    float s = policy_b[k];
    for (int j = 0; j < 1536; ++j)
      s += policy_w[(size_t)k * 1536 + j] * Pf[((size_t)b * 1536 + j) * NQ + q];
    plA[k] = s;
  }
  __syncthreads();
  if (tid == 0) {
    float z[KB];
    for (int k = 0; k < KB; ++k) {
      float u = gu[((size_t)b * KB + k) * NQ + q];
      float g = -__logf(-__logf(u + 1e-10f) + 1e-10f);
      z[k] = (plA[k] + g) / tau;
    }
    int km = 0; float zm = z[0];
    for (int k = 1; k < KB; ++k) if (z[k] > zm) { zm = z[k]; km = k; }
    for (int k = 0; k < KB; ++k) {
      maskS[k] = (k > km) ? -1e9f : 0.f;
      out[(size_t)BB * DIM * NQ + ((size_t)b * KB + k) * NQ + q] = (k == km) ? 1.f : 0.f;
    }
  }
  for (int c = tid; c < DIM; c += 256) {            // q = -(proj2 pooled)
    float s = proj2_b[c];
    for (int cc = 0; cc < 1024; ++cc)
      s += proj2_w[(size_t)c * 1024 + cc] * Pcs[((size_t)b * 1024 + cc) * NQ + q];
    qv[c] = -s;
  }
  __syncthreads();
  for (int d = tid; d < DIM; d += 256) {            // Q
    float s = qkv_b[d];
    for (int c = 0; c < DIM; ++c) s += qkv_w[(size_t)d * DIM + c] * qv[c];
    Qh[d] = s;
  }
  for (int e = tid; e < KB * DIM; e += 256) {       // K, V from nf rows
    int k = e >> 9, d = e & 511;
    const float* nfr = nf + ((size_t)bq * KB + k) * DIM;
    float sk = qkv_b[DIM + d], sv = qkv_b[2 * DIM + d];
    for (int c = 0; c < DIM; ++c) {
      float x = nfr[c];
      sk += qkv_w[((size_t)DIM + d) * DIM + c] * x;
      sv += qkv_w[((size_t)2 * DIM + d) * DIM + c] * x;
    }
    Kh[e] = sk; Vh[e] = sv;
  }
  __syncthreads();
  if (tid < NH * KB) {                              // scores = -QK/8 + mask
    int h = tid >> 2, k = tid & 3;
    float s = 0.f;
    for (int d = 0; d < 64; ++d) s += Qh[h * 64 + d] * Kh[k * DIM + h * 64 + d];
    sc[h * KB + k] = -s * 0.125f + maskS[k];
  }
  __syncthreads();
  if (tid < NH) {
    int h = tid;
    float m = sc[h * KB];
    for (int k = 1; k < KB; ++k) m = fmaxf(m, sc[h * KB + k]);
    float tot = 0.f, ex[KB];
    for (int k = 0; k < KB; ++k) { ex[k] = __expf(sc[h * KB + k] - m); tot += ex[k]; }
    for (int k = 0; k < KB; ++k) at[h * KB + k] = ex[k] / tot;
  }
  __syncthreads();
  for (int d = tid; d < DIM; d += 256) {
    int h = d >> 6;
    float s = 0.f;
    for (int k = 0; k < KB; ++k) s += at[h * KB + k] * Vh[k * DIM + d];
    out[((size_t)b * DIM + d) * NQ + q] = s;
  }
}

// ---------------------------------------------------------------------------
extern "C" void kernel_launch(void* const* d_in, const int* in_sizes, int n_in,
                              void* d_out, int out_size, void* d_ws, size_t ws_size,
                              hipStream_t stream) {
  (void)in_sizes; (void)n_in; (void)out_size; (void)ws_size;
  const float* f    = (const float*)d_in[0];
  const float* c    = (const float*)d_in[1];
  const float* s    = (const float*)d_in[2];
  const float* p1w1 = (const float*)d_in[3];
  const float* p1b1 = (const float*)d_in[4];
  const float* p1g1 = (const float*)d_in[5];
  const float* p1be1= (const float*)d_in[6];
  const float* p1w2 = (const float*)d_in[7];
  const float* p1b2 = (const float*)d_in[8];
  const float* fw1  = (const float*)d_in[9];
  const float* fb1  = (const float*)d_in[10];
  const float* fg1  = (const float*)d_in[11];
  const float* fbe1 = (const float*)d_in[12];
  const float* fw2  = (const float*)d_in[13];
  const float* fb2  = (const float*)d_in[14];
  const float* fg2  = (const float*)d_in[15];
  const float* fbe2 = (const float*)d_in[16];
  const float* qkvw = (const float*)d_in[17];
  const float* qkvb = (const float*)d_in[18];
  const float* p2w  = (const float*)d_in[19];
  const float* p2b  = (const float*)d_in[20];
  const float* polw = (const float*)d_in[21];
  const float* polb = (const float*)d_in[22];
  const float* gu   = (const float*)d_in[23];
  const int*   tau  = (const int*)d_in[24];

  char* ws = (char*)d_ws;
  size_t off = 0;
  auto alloc = [&](size_t bytes) {
    void* p = ws + off; off = (off + bytes + 255) & ~(size_t)255; return p;
  };
  _Float16* packA = (_Float16*)alloc((size_t)27 * MATH * 2);
  float*    att   = (float*)   alloc((size_t)BB * 3 * KB * TT * 4);
  _Float16* nk    = (_Float16*)alloc((size_t)KB * BB * DIM * TT * 2);
  _Float16* h1    = (_Float16*)alloc((size_t)KB * BB * DIM * TT * 2);
  float*    nf    = (float*)   alloc((size_t)BB * NQ * KB * DIM * 4);
  float*    Pcs   = (float*)   alloc((size_t)BB * 1024 * NQ * 4);
  float*    Pf    = (float*)   alloc((size_t)BB * 1536 * NQ * 4);
  float*    bn    = (float*)   alloc(11264 * 4);

  k_prep_bn<<<22, 256, 0, stream>>>(p1g1, p1be1, p1b1, fg1, fbe1, fb1,
                                    fg2, fbe2, fb2, bn);
  for (int i = 0; i < 3; ++i)
    k_pack<<<1024, 256, 0, stream>>>(p1w1 + (size_t)i * DIM * DIM, DIM, 1,
                                     packA + (size_t)i * MATH);
  for (int k = 0; k < KB; ++k)
    for (int tap = 0; tap < 3; ++tap) {
      k_pack<<<1024, 256, 0, stream>>>(fw1 + (size_t)k * DIM * DIM * 3 + tap,
                                       DIM * 3, 3,
                                       packA + (size_t)(3 + k * 3 + tap) * MATH);
      k_pack<<<1024, 256, 0, stream>>>(fw2 + (size_t)k * DIM * DIM * 3 + tap,
                                       DIM * 3, 3,
                                       packA + (size_t)(15 + k * 3 + tap) * MATH);
    }
  k_proj1_att<<<512, 256, 0, stream>>>(f, packA, p1w2, p1b2, bn, att);
  k_nk<<<dim3(16, 4, 16), 256, 0, stream>>>(f, att, nk, nf);
  k_conv3<<<dim3(512, 4), 256, 0, stream>>>(nk, packA, 3,  bn, 3072, h1, nf, 0);
  k_conv3<<<dim3(512, 4), 256, 0, stream>>>(h1, packA, 15, bn, 7168, nullptr, nf, 1);
  k_pool_cs<<<64, 256, 0, stream>>>(c, s, Pcs);
  k_pool_f<<<64, 256, 0, stream>>>(f, Pf);
  k_final<<<64, 256, 0, stream>>>(Pcs, Pf, nf, p2w, p2b, polw, polb,
                                  qkvw, qkvb, gu, tau, (float*)d_out);
}
